// Nonlocal_88527865905827
// MI455X (gfx1250) — matrix operand order repaired and checked by reading, not compiled, not run on hardware
//
#include <hip/hip_runtime.h>
#include <hip/hip_bf16.h>

// ---------------------------------------------------------------------------
// Nonlocal block for MI455X (gfx1250, wave32, WMMA bf16 16x16x32 + TDM).
// b=8, C=256, IC=64, xp: 64x64 (N=4096), pooled g/phi: 32x32 (M=1024).
// All WMMA operands are loaded with K contiguous per lane -> b128 loads.
// k_attn stages the per-batch phi/g slabs into LDS via TENSOR_LOAD_TO_LDS.
// ---------------------------------------------------------------------------

typedef __attribute__((ext_vector_type(16))) __bf16       v16bf;
typedef __attribute__((ext_vector_type(8)))  float        v8f;
typedef __attribute__((ext_vector_type(4)))  unsigned int u32x4;
typedef __attribute__((ext_vector_type(8)))  unsigned int u32x8;
typedef __attribute__((ext_vector_type(4)))  int          i32x4;
typedef __attribute__((ext_vector_type(8)))  int          i32x8;

#define B_  8
#define C_  256
#define IC_ 64
#define N_  4096   // 64*64
#define M_  1024   // 32*32

static __device__ inline v8f wmma_bf16(v16bf a, v16bf b, v8f c) {
  return __builtin_amdgcn_wmma_f32_16x16x32_bf16(false, a, false, b, (short)0, c,
                                                 false, false);
}

// A-tile fragment (16-bit A layout, ISA 7.12.2):
//  lane: row = lane&15, kh = lane>>4; elems 0..7 <- p[0..7]  (K = kb+8*kh+j)
//  elems 8..15 <- p[16..23] (K = kb+16+8*kh+j).  p must be 16B aligned.
static __device__ inline v16bf load16bf_split(const __bf16* p) {
  u32x4 a = *(const u32x4*)p;
  u32x4 b = *(const u32x4*)(p + 16);
  u32x8 t;
  t[0] = a[0]; t[1] = a[1]; t[2] = a[2]; t[3] = a[3];
  t[4] = b[0]; t[5] = b[1]; t[6] = b[2]; t[7] = b[3];
  return __builtin_bit_cast(v16bf, t);
}

// B-tile fragment (this code's contiguous K order): lane: n = lane&15, kh = lane>>4,
//  elem e <- p[e]  (K = kb + 16*kh + e).  16 contiguous bf16 = 32 bytes.
static __device__ inline v16bf load16bf_contig(const __bf16* p) {
  u32x4 a = *(const u32x4*)p;
  u32x4 b = *(const u32x4*)(p + 8);
  u32x8 t;
  t[0] = a[0]; t[1] = a[1]; t[2] = a[2]; t[3] = a[3];
  t[4] = b[0]; t[5] = b[1]; t[6] = b[2]; t[7] = b[3];
  return __builtin_bit_cast(v16bf, t);
}

// A: source row-major bf16, rowLane = row0 + (lane&15), K contiguous in row.
static __device__ inline v16bf load_A16(const __bf16* __restrict__ m, int rowLane,
                                        int ld, int kb, int kh) {
  return load16bf_contig(m + (size_t)rowLane * ld + kb + kh * 16);
}
// B: source has B-columns as rows (K contiguous): B[k][n] = m[rowLane*ld + kb + k]
static __device__ inline v16bf load_B16(const __bf16* __restrict__ m, int rowLane,
                                        int ld, int kb, int kh) {
  return load16bf_contig(m + (size_t)rowLane * ld + kb + kh * 16);
}

// ---------------------------------------------------------------------------
// TDM: issue a 2D TENSOR_LOAD_TO_LDS (data_size = 2 bytes).
// D# packing per CDNA5 ISA ch.8: group0 = {count, lds_addr, global_addr, type=2},
// group1 = {mask/data_size, tensor_dim0/1, tile_dim0/1, dim0_stride}.
// Groups 2/3 zero (tensor is 2D).  Tracked by TENSORcnt.
// ---------------------------------------------------------------------------
static __device__ inline void tdm_load_2d_bf16(unsigned lds_addr, const void* gptr,
                                               unsigned tensor_d0, unsigned tensor_d1,
                                               unsigned tile_d0, unsigned tile_d1,
                                               unsigned stride0_elems) {
  unsigned long long ga = (unsigned long long)(size_t)gptr;
  u32x4 g0;
  g0[0] = 1u;                                        // count=1 (valid), user mode
  g0[1] = lds_addr;                                  // LDS byte address
  g0[2] = (unsigned)(ga & 0xffffffffu);              // global_addr[31:0]
  g0[3] = (unsigned)((ga >> 32) & 0x01ffffffu)       // global_addr[56:32]
          | (2u << 30);                              // type = 2 ("image")
  i32x8 g1;
  g1[0] = (int)(1u << 16);                           // wg_mask=0, data_size=1 (2B)
  g1[1] = (int)((tensor_d0 & 0xffffu) << 16);        // tensor_dim0[15:0]
  g1[2] = (int)(((tensor_d0 >> 16) & 0xffffu) |
                ((tensor_d1 & 0xffffu) << 16));      // td0[31:16] | td1[15:0]
  g1[3] = (int)(((tensor_d1 >> 16) & 0xffffu) |
                ((tile_d0 & 0xffffu) << 16));        // td1[31:16] | tile_d0
  g1[4] = (int)(tile_d1 & 0xffffu);                  // tile_d1, tile_d2=0
  g1[5] = (int)stride0_elems;                        // tensor_dim0_stride[31:0]
  g1[6] = 0;                                         // stride0[47:32]=0, s1 lo=0
  g1[7] = 0;                                         // tensor_dim1_stride hi = 0
  i32x4 z4 = {0, 0, 0, 0};
#if __clang_major__ >= 23
  i32x8 z8 = {0, 0, 0, 0, 0, 0, 0, 0};
  __builtin_amdgcn_tensor_load_to_lds(g0, g1, z4, z4, z8, 0);
#else
  __builtin_amdgcn_tensor_load_to_lds(g0, g1, z4, z4, 0);
#endif
}

// ---------------------------------------------------------------------------
// K0: convert the four weight matrices to bf16 (once; 16384 elems each)
// ---------------------------------------------------------------------------
__global__ void k_wcvt(const float* __restrict__ tw, const float* __restrict__ gw,
                       const float* __restrict__ pw, const float* __restrict__ Ww,
                       __bf16* __restrict__ twb, __bf16* __restrict__ gwb,
                       __bf16* __restrict__ pwb, __bf16* __restrict__ Wwb) {
  int i = blockIdx.x * 256 + threadIdx.x;   // 16384 total
  twb[i] = (__bf16)tw[i];
  gwb[i] = (__bf16)gw[i];
  pwb[i] = (__bf16)pw[i];
  Wwb[i] = (__bf16)Ww[i];
}

// ---------------------------------------------------------------------------
// K1: 2x2 maxpool of x -> xp (f32)
// ---------------------------------------------------------------------------
__global__ void k_pool_x(const float* __restrict__ x, float* __restrict__ xp) {
  int idx = blockIdx.x * 256 + threadIdx.x;       // b*C*4096 total
  int pix = idx & 4095;
  int bc  = idx >> 12;
  int oy = pix >> 6, ox = pix & 63;
  const float* src = x + ((size_t)bc << 14);      // 128*128 per plane
  int i0 = (oy * 2) * 128 + ox * 2;
  float a = src[i0], b = src[i0 + 1], c = src[i0 + 128], d = src[i0 + 129];
  xp[idx] = fmaxf(fmaxf(a, b), fmaxf(c, d));
}

// ---------------------------------------------------------------------------
// K1t: LDS-tiled transpose xp (b,C,N) f32 -> xpT (b,N,C) bf16
// ---------------------------------------------------------------------------
__global__ void k_tr(const float* __restrict__ xp, __bf16* __restrict__ xpT) {
  __shared__ __bf16 s[64][65];
  int bid = blockIdx.x;
  int b = bid >> 8; int rem = bid & 255; int ct = rem >> 6; int pt = rem & 63;
  int c0 = ct * 64, p0 = pt * 64;
  const float* src = xp + ((size_t)b * C_ + c0) * N_ + p0;
  for (int i = threadIdx.x; i < 4096; i += 256) {
    int cc = i >> 6, pp = i & 63;                 // coalesced reads along pix
    s[pp][cc] = (__bf16)src[(size_t)cc * N_ + pp];
  }
  __syncthreads();
  __bf16* dst = xpT + ((size_t)b * N_ + p0) * C_ + c0;
  for (int i = threadIdx.x; i < 4096; i += 256) {
    int pp = i >> 6, cc = i & 63;                 // coalesced writes along c
    dst[(size_t)pp * C_ + cc] = s[pp][cc];
  }
}

// ---------------------------------------------------------------------------
// K1b: 16x16 maxpool of ndwi (8,1024,1024) -> nd (8,4096)
// ---------------------------------------------------------------------------
__global__ void k_pool_nd(const float* __restrict__ ndwi, float* __restrict__ nd) {
  __shared__ float sm[256];
  int tid = threadIdx.x;
  const float* src = ndwi + ((size_t)blockIdx.z << 20) +
                     (size_t)(blockIdx.y * 16) * 1024 + blockIdx.x * 256;
  float m = -3.0e38f;
#pragma unroll
  for (int r = 0; r < 16; ++r) m = fmaxf(m, src[r * 1024 + tid]);
  sm[tid] = m;
  __syncthreads();
  for (int s = 8; s; s >>= 1) {
    if ((tid & 15) < s) sm[tid] = fmaxf(sm[tid], sm[tid + s]);
    __syncthreads();
  }
  if ((tid & 15) == 0)
    nd[blockIdx.z * 4096 + blockIdx.y * 64 + blockIdx.x * 16 + (tid >> 4)] = sm[tid];
}

// ---------------------------------------------------------------------------
// K2: fused theta/g/phi 1x1 convs as WMMA GEMMs over xpT (b,N,C) bf16.
//  theta -> (b, N, IC) bf16
//  g     -> (b, IC, M) bf16  (conv then 2x2 maxpool, 4 phases fused)
//  phi   -> (b, M, IC) bf16
// One wave per 16x16 output tile. 12288 wave-jobs, 8 waves/block.
// ---------------------------------------------------------------------------
__global__ void k_qkv(const __bf16* __restrict__ xpT,
                      const __bf16* __restrict__ twb, const float* __restrict__ tb,
                      const __bf16* __restrict__ gwb, const float* __restrict__ gbi,
                      const __bf16* __restrict__ pwb, const float* __restrict__ pb,
                      __bf16* __restrict__ thetaO, __bf16* __restrict__ gO,
                      __bf16* __restrict__ phiO) {
  int lane = threadIdx.x & 31;
  int wv   = threadIdx.x >> 5;
  int jid  = blockIdx.x * 8 + wv;   // 0..12287, wave-uniform
  int kh = lane >> 4, nn = lane & 15;

  if (jid < 8192) {
    // ---- theta: full-resolution conv ----
    int b = jid >> 10; int rem = jid & 1023; int ict = rem >> 8; int pt = rem & 255;
    int ic0 = ict * 16, n0 = pt * 16;
    const __bf16* xb = xpT + (size_t)b * N_ * C_;
    v8f acc = {};
    for (int kk = 0; kk < C_; kk += 32) {
      v16bf a  = load_A16(twb, ic0 + nn, C_, kk, kh);
      v16bf bm = load_B16(xb, n0 + nn, C_, kk, kh);   // B[k][n] = xpT[pix][c]
      acc = wmma_bf16(a, bm, acc);
    }
#pragma unroll
    for (int r = 0; r < 8; ++r) {
      int ic = ic0 + r + kh * 8;
      int n  = n0 + nn;
      thetaO[((size_t)b * N_ + n) * IC_ + ic] = (__bf16)(acc[r] + tb[ic]);
    }
  } else {
    // ---- g / phi: conv + fused 2x2 maxpool (4 phase GEMMs, shared A) ----
    int jid2 = jid - 8192;
    bool isG = jid2 < 2048;
    int j = isG ? jid2 : jid2 - 2048;
    int b = j >> 8; int rem = j & 255; int ict = rem >> 6; int mt = rem & 63;
    int ic0 = ict * 16, m0 = mt * 16;
    int my = m0 >> 5, mx0 = m0 & 31;    // pooled tile is one row of 16 (16|32)
    const __bf16* w   = isG ? gwb : pwb;
    const float* bias = isG ? gbi : pb;
    const __bf16* xb = xpT + (size_t)b * N_ * C_;
    v8f acc[4] = {{}, {}, {}, {}};
    for (int kk = 0; kk < C_; kk += 32) {
      v16bf a = load_A16(w, ic0 + nn, C_, kk, kh);
#pragma unroll
      for (int ph = 0; ph < 4; ++ph) {
        int dy = ph >> 1, dx = ph & 1;
        int pix = (2 * my + dy) * 64 + 2 * (mx0 + nn) + dx;
        v16bf bm = load_B16(xb, pix, C_, kk, kh);
        acc[ph] = wmma_bf16(a, bm, acc[ph]);
      }
    }
#pragma unroll
    for (int r = 0; r < 8; ++r) {
      float v = fmaxf(fmaxf(acc[0][r], acc[1][r]), fmaxf(acc[2][r], acc[3][r]));
      int ic = ic0 + r + kh * 8;
      int m  = m0 + nn;
      v += bias[ic];
      if (isG) gO[((size_t)b * IC_ + ic) * M_ + m]   = (__bf16)v;
      else     phiO[((size_t)b * M_ + m) * IC_ + ic] = (__bf16)v;
    }
  }
}

// ---------------------------------------------------------------------------
// K3: flash-attention pass.  8 waves/block, all on the SAME batch, so the
// per-batch phi (1024x64) and g (64x1024) bf16 slabs (128KB each) are staged
// into LDS once per block via TENSOR_LOAD_TO_LDS (TDM), then every B-fragment
// is a pair of ds_load_b128.  Dynamic LDS: 256KB slabs + 8KB P-relayout.
// ---------------------------------------------------------------------------
__global__ void k_attn(const __bf16* __restrict__ theta,
                       const __bf16* __restrict__ phi,
                       const __bf16* __restrict__ g,
                       const float* __restrict__ nd,
                       __bf16* __restrict__ yO) {
  extern __shared__ char dsm[];
  __bf16* lphi = (__bf16*)dsm;                    // 1024 x 64
  __bf16* lg   = (__bf16*)(dsm + 131072);         // 64 x 1024
  __bf16* spb  = (__bf16*)(dsm + 262144);         // 8 waves x 512 (P relayout)

  int lane = threadIdx.x & 31, wv = threadIdx.x >> 5;
  int kh = lane >> 4, nn = lane & 15;
  int tile = blockIdx.x * 8 + wv;     // 0..2047
  int b = tile >> 8;                  // all waves in a block share b
  int n0 = (tile & 255) * 16;

  const __bf16* thb = theta + (size_t)b * N_ * IC_;
  const __bf16* phb = phi   + (size_t)b * M_ * IC_;
  const __bf16* gb_ = g     + (size_t)b * IC_ * M_;

  // ---- TDM stage: one wave issues both slab copies, waits TENSORcnt ----
  unsigned lds0 = (unsigned)__builtin_amdgcn_groupstaticsize();
  if (wv == 0) {
    tdm_load_2d_bf16(lds0,          phb, 64, 1024, 64, 1024, 64);    // phi slab
    tdm_load_2d_bf16(lds0 + 131072, gb_, 1024, 64, 1024, 64, 1024);  // g slab
    __builtin_amdgcn_s_wait_tensorcnt(0);
  }
  __syncthreads();                     // publish LDS slabs to all waves

  v16bf at0 = load_A16(thb, n0 + nn, IC_, 0, kh);   // K = ic 0..31
  v16bf at1 = load_A16(thb, n0 + nn, IC_, 32, kh);  // K = ic 32..63

  v8f yacc[4] = {{}, {}, {}, {}};
  float rm[8], rl[8];
#pragma unroll
  for (int r = 0; r < 8; ++r) { rm[r] = -1.0e30f; rl[r] = 0.0f; }

  __bf16* myp = spb + wv * 512;

  for (int mc = 0; mc < M_; mc += 32) {
    v8f s0 = {}, s1 = {};
    s0 = wmma_bf16(at0, load_B16(lphi, mc + nn, IC_, 0, kh), s0);
    s0 = wmma_bf16(at1, load_B16(lphi, mc + nn, IC_, 32, kh), s0);
    s1 = wmma_bf16(at0, load_B16(lphi, mc + 16 + nn, IC_, 0, kh), s1);
    s1 = wmma_bf16(at1, load_B16(lphi, mc + 16 + nn, IC_, 32, kh), s1);

    float p0[8], p1[8];
#pragma unroll
    for (int r = 0; r < 8; ++r) {
      float mx = fmaxf(s0[r], s1[r]);
      for (int s = 8; s; s >>= 1) mx = fmaxf(mx, __shfl_xor(mx, s, 16));
      float mnew  = fmaxf(rm[r], mx);
      float alpha = __expf(rm[r] - mnew);
      float e0 = __expf(s0[r] - mnew);
      float e1 = __expf(s1[r] - mnew);
      float rs = e0 + e1;
      for (int s = 8; s; s >>= 1) rs += __shfl_xor(rs, s, 16);
      rl[r] = rl[r] * alpha + rs;
      rm[r] = mnew;
      p0[r] = e0; p1[r] = e1;
#pragma unroll
      for (int t = 0; t < 4; ++t) yacc[t][r] *= alpha;
    }

    // C-layout (s0/s1) -> A-layout bf16 for P via LDS bounce
    __syncthreads();
#pragma unroll
    for (int r = 0; r < 8; ++r) {
      int row = r + kh * 8;
      myp[row * 32 + nn]      = (__bf16)p0[r];
      myp[row * 32 + 16 + nn] = (__bf16)p1[r];
    }
    __syncthreads();
    v16bf pa = load16bf_contig(myp + nn * 32 + kh * 16);   // 2x ds_load_b128

#pragma unroll
    for (int t = 0; t < 4; ++t) {
      v16bf bg = load_B16(lg, t * 16 + nn, M_, mc, kh);  // B[k][ic] = g[ic][mc+k]
      yacc[t] = wmma_bf16(pa, bg, yacc[t]);
    }
  }

#pragma unroll
  for (int r = 0; r < 8; ++r) {
    int n = n0 + r + kh * 8;
    float sc = nd[b * N_ + n] / rl[r];
#pragma unroll
    for (int t = 0; t < 4; ++t)
      yO[((size_t)b * N_ + n) * IC_ + t * 16 + nn] = (__bf16)(yacc[t][r] * sc);
  }
}

// ---------------------------------------------------------------------------
// K4: W conv (64->256) via WMMA + deterministic BN partials.
// ---------------------------------------------------------------------------
__global__ void k_wconv(const __bf16* __restrict__ y,
                        const __bf16* __restrict__ Wwb, const float* __restrict__ Wb,
                        float* __restrict__ Wy,
                        float* __restrict__ psum, float* __restrict__ psq) {
  __shared__ float ssum[8][16], ssq[8][16];
  int lane = threadIdx.x & 31, wv = threadIdx.x >> 5;
  int kh = lane >> 4, nn = lane & 15;
  int bid = blockIdx.x;
  int b = bid >> 9; int rem = bid & 511; int ct = rem >> 5; int nt = rem & 31;
  int c0 = ct * 16;
  int n0 = nt * 128 + wv * 16;
  const __bf16* yb = y + (size_t)b * N_ * IC_;

  v8f acc = {};
  acc = wmma_bf16(load_A16(Wwb, c0 + nn, IC_, 0, kh),
                  load_B16(yb, n0 + nn, IC_, 0, kh), acc);
  acc = wmma_bf16(load_A16(Wwb, c0 + nn, IC_, 32, kh),
                  load_B16(yb, n0 + nn, IC_, 32, kh), acc);

  float lsum[8], lsq[8];
#pragma unroll
  for (int r = 0; r < 8; ++r) {
    int c = c0 + r + kh * 8;
    float v = acc[r] + Wb[c];
    Wy[((size_t)b * C_ + c) * N_ + n0 + nn] = v;
    float s = v, q = v * v;
    for (int t = 8; t; t >>= 1) { s += __shfl_xor(s, t, 16); q += __shfl_xor(q, t, 16); }
    lsum[r] = s; lsq[r] = q;
  }
  if (nn == 0) {
#pragma unroll
    for (int r = 0; r < 8; ++r) {
      ssum[wv][r + kh * 8] = lsum[r];
      ssq[wv][r + kh * 8]  = lsq[r];
    }
  }
  __syncthreads();
  int tid = threadIdx.x;
  if (tid < 16) {
    float s = 0.f, q = 0.f;
    for (int w = 0; w < 8; ++w) { s += ssum[w][tid]; q += ssq[w][tid]; }
    int c = c0 + tid;
    int pidx = b * 32 + nt;
    psum[c * 256 + pidx] = s;
    psq[c * 256 + pidx]  = q;
  }
}

// K4b: fold batch stats into per-channel scale/shift (deterministic order)
__global__ void k_bnfin(const float* __restrict__ psum, const float* __restrict__ psq,
                        const float* __restrict__ gamma, const float* __restrict__ beta,
                        float* __restrict__ scale, float* __restrict__ shift) {
  int c = threadIdx.x;
  float s = 0.f, q = 0.f;
  for (int j = 0; j < 256; ++j) { s += psum[c * 256 + j]; q += psq[c * 256 + j]; }
  const float inv = 1.0f / (float)(B_ * N_);
  float mean = s * inv;
  float var  = q * inv - mean * mean;
  float sc = gamma[c] * rsqrtf(var + 1e-5f);
  scale[c] = sc;
  shift[c] = beta[c] - mean * sc;
}

// ---------------------------------------------------------------------------
// K5: z = BN(Wy) + xp staged in LDS, then 2x bilinear upsample (align_corners)
// ---------------------------------------------------------------------------
__global__ void k_up(const float* __restrict__ Wy, const float* __restrict__ xp,
                     const float* __restrict__ scale, const float* __restrict__ shift,
                     float* __restrict__ out) {
  __shared__ float zt[4096];
  int plane = blockIdx.x;               // b*256 + c
  int c = plane & 255;
  float sc = scale[c], sh = shift[c];
  const float* wy = Wy + (size_t)plane * 4096;
  const float* xb = xp + (size_t)plane * 4096;
  for (int i = threadIdx.x; i < 4096; i += 256)
    zt[i] = fmaf(wy[i], sc, sh) + xb[i];
  __syncthreads();
  float* ob = out + (size_t)plane * 16384;
  const float rr = 63.0f / 127.0f;
  for (int i = threadIdx.x; i < 16384; i += 256) {
    int oy = i >> 7, ox = i & 127;
    float fy = oy * rr, fx = ox * rr;
    int y0 = (int)fy; float wyf = fy - (float)y0; int y1 = min(y0 + 1, 63);
    int x0 = (int)fx; float wxf = fx - (float)x0; int x1 = min(x0 + 1, 63);
    float t0 = zt[y0 * 64 + x0], t1 = zt[y0 * 64 + x1];
    float b0 = zt[y1 * 64 + x0], b1 = zt[y1 * 64 + x1];
    float top = t0 + (t1 - t0) * wxf;
    float bot = b0 + (b1 - b0) * wxf;
    ob[i] = top + (bot - top) * wyf;
  }
}

// ---------------------------------------------------------------------------
extern "C" void kernel_launch(void* const* d_in, const int* in_sizes, int n_in,
                              void* d_out, int out_size, void* d_ws, size_t ws_size,
                              hipStream_t stream) {
  const float* x     = (const float*)d_in[0];
  const float* ndwi  = (const float*)d_in[1];
  const float* g_w   = (const float*)d_in[2];
  const float* g_b   = (const float*)d_in[3];
  const float* th_w  = (const float*)d_in[4];
  const float* th_b  = (const float*)d_in[5];
  const float* ph_w  = (const float*)d_in[6];
  const float* ph_b  = (const float*)d_in[7];
  const float* W_w   = (const float*)d_in[8];
  const float* W_b   = (const float*)d_in[9];
  const float* gamma = (const float*)d_in[10];
  const float* beta  = (const float*)d_in[11];
  float* out = (float*)d_out;

  char* ws = (char*)d_ws;
  size_t off = 0;
  auto alloc = [&](size_t bytes) -> char* {
    char* p = ws + off;
    off += (bytes + 255) & ~(size_t)255;
    return p;
  };
  float*  xp    = (float*)alloc(sizeof(float) * (size_t)B_ * C_ * N_);   // 33.5MB
  float*  Wy    = (float*)alloc(sizeof(float) * (size_t)B_ * C_ * N_);   // 33.5MB
  __bf16* xpT   = (__bf16*)alloc(2ull * B_ * N_ * C_);                   // 16.8MB
  __bf16* theta = (__bf16*)alloc(2ull * B_ * N_ * IC_);                  // 4.2MB
  __bf16* yb    = (__bf16*)alloc(2ull * B_ * N_ * IC_);                  // 4.2MB
  __bf16* phi   = (__bf16*)alloc(2ull * B_ * M_ * IC_);                  // 1.05MB
  __bf16* gbuf  = (__bf16*)alloc(2ull * B_ * IC_ * M_);                  // 1.05MB
  __bf16* twb   = (__bf16*)alloc(2ull * IC_ * C_);
  __bf16* gwb   = (__bf16*)alloc(2ull * IC_ * C_);
  __bf16* pwb   = (__bf16*)alloc(2ull * IC_ * C_);
  __bf16* Wwb   = (__bf16*)alloc(2ull * C_ * IC_);
  float*  nd    = (float*)alloc(sizeof(float) * (size_t)B_ * N_);
  float*  psum  = (float*)alloc(sizeof(float) * 256 * 256);
  float*  psq   = (float*)alloc(sizeof(float) * 256 * 256);
  float*  scale = (float*)alloc(sizeof(float) * 256);
  float*  shift = (float*)alloc(sizeof(float) * 256);

  k_wcvt<<<64, 256, 0, stream>>>(th_w, g_w, ph_w, W_w, twb, gwb, pwb, Wwb);
  k_pool_x<<<32768, 256, 0, stream>>>(x, xp);
  k_tr<<<2048, 256, 0, stream>>>(xp, xpT);
  k_pool_nd<<<dim3(4, 64, 8), 256, 0, stream>>>(ndwi, nd);
  k_qkv<<<1536, 256, 0, stream>>>(xpT, twb, th_b, gwb, g_b, pwb, ph_b,
                                  theta, gbuf, phi);
  // dynamic LDS: 2 x 128KB slabs + 8 x 1KB P-relayout = 270336 bytes
  k_attn<<<256, 256, 270336, stream>>>(theta, phi, gbuf, nd, yb);
  k_wconv<<<4096, 256, 0, stream>>>(yb, Wwb, W_b, Wy, psum, psq);
  k_bnfin<<<1, 256, 0, stream>>>(psum, psq, gamma, beta, scale, shift);
  k_up<<<2048, 256, 0, stream>>>(Wy, xp, scale, shift, out);
}
